// NTXentLoss_66709432041521
// MI455X (gfx1250) — matrix operand order repaired and checked by reading, not compiled, not run on hardware
//
#include <hip/hip_runtime.h>

#define NROWS 8192      // 2B
#define BSZ   4096      // B
#define DDIM  128
#define TILES 512       // NROWS / 16
#define NEGW  8190u     // 2B - 2

typedef __attribute__((ext_vector_type(16))) _Float16 v16h;
typedef __attribute__((ext_vector_type(8)))  _Float16 v8h;
typedef __attribute__((ext_vector_type(4)))  _Float16 v4h;
typedef __attribute__((ext_vector_type(8)))  float    v8f;

// ---------------------------------------------------------------------------
// Zero the per-logits-row accumulators and the output scalar (harness poisons
// buffers and does not re-zero between replays; atomics require clean state).
// ---------------------------------------------------------------------------
__global__ void ntx_zero_kernel(float* __restrict__ accum, float* __restrict__ out) {
    int idx = blockIdx.x * blockDim.x + threadIdx.x;
    if (idx < BSZ) accum[idx] = 0.0f;
    if (idx == 0)  out[0] = 0.0f;
}

// ---------------------------------------------------------------------------
// Row-normalize reps = concat(zjs, zis). One wave32 per row; each lane owns 4
// contiguous floats. Writes fp32 copy (for exact positive dots) and f16 copy
// (for WMMA).
// ---------------------------------------------------------------------------
__global__ void ntx_normalize_kernel(const float* __restrict__ zis,
                                     const float* __restrict__ zjs,
                                     float* __restrict__ nF,
                                     _Float16* __restrict__ nH) {
    int wave = threadIdx.x >> 5;
    int lane = threadIdx.x & 31;
    int row  = blockIdx.x * 8 + wave;

    const float* src = (row < BSZ) ? (zjs + (size_t)row * DDIM)
                                   : (zis + (size_t)(row - BSZ) * DDIM);
    float4 v = *(const float4*)(src + lane * 4);
    float ss = v.x * v.x + v.y * v.y + v.z * v.z + v.w * v.w;
    #pragma unroll
    for (int off = 16; off > 0; off >>= 1) ss += __shfl_xor(ss, off, 32);

    float scale = 1.0f / fmaxf(sqrtf(ss), 1e-8f);
    float4 o = make_float4(v.x * scale, v.y * scale, v.z * scale, v.w * scale);
    *(float4*)(nF + (size_t)row * DDIM + lane * 4) = o;

    v4h h;
    h[0] = (_Float16)o.x; h[1] = (_Float16)o.y;
    h[2] = (_Float16)o.z; h[3] = (_Float16)o.w;
    *(v4h*)(nH + (size_t)row * DDIM + lane * 4) = h;
}

// ---------------------------------------------------------------------------
// One wave per 16x16 tile of sim = N @ N^T, upper triangle only (ti <= tj).
// K = 128 via 4 chained v_wmma_f32_16x16x32_f16. Each C element (i,j) with
// j > i (and not the positive) contributes exp(2*sim) to accum[k], where k is
// the logits row given by the reference's row-major negative-stream reshape.
// ---------------------------------------------------------------------------
__global__ void ntx_simtile_kernel(const _Float16* __restrict__ nH,
                                   float* __restrict__ accum) {
    int tj = blockIdx.x;
    int ti = blockIdx.y;
    if (ti > tj) return;   // uniform across the wave: EXEC stays all-ones

    int lane = threadIdx.x;            // 0..31
    int g    = lane >> 4;              // half-wave group
    int sub  = lane & 15;              // A: row m, B: col n

    const _Float16* Arow = nH + (size_t)(ti * 16 + sub) * DDIM;
    const _Float16* Brow = nH + (size_t)(tj * 16 + sub) * DDIM;

    v8f c = {};
    #pragma unroll
    for (int kk = 0; kk < 4; ++kk) {
        int koff = kk * 32;
        // A 16x32 f16 layout: lane group g holds K = g*8..g*8+7 and 16+g*8..
        v8h a0 = *(const v8h*)(Arow + koff + g * 8);
        v8h a1 = *(const v8h*)(Arow + koff + 16 + g * 8);
        v16h a = __builtin_shufflevector(a0, a1,
                 0,1,2,3,4,5,6,7,8,9,10,11,12,13,14,15);
        // B 32x16 f16 layout: lane group g holds K = g*8..g*8+7 and 16+g*8.., col = sub
        v8h b0 = *(const v8h*)(Brow + koff + g * 8);
        v8h b1 = *(const v8h*)(Brow + koff + 16 + g * 8);
        v16h b = __builtin_shufflevector(b0, b1,
                 0,1,2,3,4,5,6,7,8,9,10,11,12,13,14,15);
        c = __builtin_amdgcn_wmma_f32_16x16x32_f16(
                false, a, false, b, (short)0, c, false, false);
    }

    int j = tj * 16 + sub;             // C/D layout: N = lane%16
    #pragma unroll
    for (int v = 0; v < 8; ++v) {
        int i = ti * 16 + v + 8 * g;   // C/D layout: M = v + 8*(lane/16)
        if (j > i) {
            bool overPos = (i < BSZ) && (j > i + BSZ);
            bool isPos   = (i < BSZ) && (j == i + BSZ);
            if (!isPos) {
                unsigned ui = (unsigned)i;
                // prefix count of negatives before row i in the flat stream
                unsigned S = ui * (unsigned)(NROWS - 1)
                           - (ui * (ui - 1u)) / 2u
                           - (ui < BSZ ? ui : (unsigned)BSZ);
                unsigned r = (unsigned)(j - i - 1) - (overPos ? 1u : 0u);
                unsigned k = (S + r) / NEGW;   // logits row
                atomicAdd(&accum[k], __expf(c[v] * 2.0f));
            }
        }
    }
}

// ---------------------------------------------------------------------------
// One wave per logits row k: exact fp32 positive dot, then
// contrib = (log(negsum + exp(pos)) - pos) / B, summed into out.
// ---------------------------------------------------------------------------
__global__ void ntx_finalize_kernel(const float* __restrict__ nF,
                                    const float* __restrict__ accum,
                                    float* __restrict__ out) {
    int wave = threadIdx.x >> 5;
    int lane = threadIdx.x & 31;
    int k = blockIdx.x * 8 + wave;

    float4 a = *(const float4*)(nF + (size_t)k * DDIM + lane * 4);
    float4 b = *(const float4*)(nF + (size_t)(k + BSZ) * DDIM + lane * 4);
    float d = a.x * b.x + a.y * b.y + a.z * b.z + a.w * b.w;
    #pragma unroll
    for (int off = 16; off > 0; off >>= 1) d += __shfl_xor(d, off, 32);

    if (lane == 0) {
        float posl  = 2.0f * d;                       // positive logit (T=0.5)
        float total = accum[k] + __expf(posl);        // full logsumexp arg
        float contrib = (__logf(total) - posl) * (1.0f / (float)BSZ);
        atomicAdd(out, contrib);
    }
}

extern "C" void kernel_launch(void* const* d_in, const int* in_sizes, int n_in,
                              void* d_out, int out_size, void* d_ws, size_t ws_size,
                              hipStream_t stream) {
    (void)in_sizes; (void)n_in; (void)out_size; (void)ws_size;

    const float* zis = (const float*)d_in[0];
    const float* zjs = (const float*)d_in[1];
    float* out = (float*)d_out;

    char* ws = (char*)d_ws;
    float*    nF    = (float*)ws;                                   // 4 MiB
    _Float16* nH    = (_Float16*)(ws + (size_t)NROWS * DDIM * 4);   // 2 MiB
    float*    accum = (float*)(ws + (size_t)NROWS * DDIM * 4
                                  + (size_t)NROWS * DDIM * 2);      // 16 KiB

    ntx_zero_kernel<<<dim3((BSZ + 255) / 256), dim3(256), 0, stream>>>(accum, out);
    ntx_normalize_kernel<<<dim3(NROWS / 8), dim3(256), 0, stream>>>(zis, zjs, nF, nH);
    ntx_simtile_kernel<<<dim3(TILES, TILES), dim3(32), 0, stream>>>(nH, accum);
    ntx_finalize_kernel<<<dim3(BSZ / 8), dim3(256), 0, stream>>>(nF, accum, out);
}
